// PointerAttentionDecoder_58720792871425
// MI455X (gfx1250) — matrix operand order repaired and checked by reading, not compiled, not run on hardware
//
#include <hip/hip_runtime.h>
#include <math.h>

typedef __attribute__((ext_vector_type(16))) __bf16 bf16x16;
typedef __attribute__((ext_vector_type(8)))  __bf16 bf16x8;
typedef __attribute__((ext_vector_type(8)))  float  f32x8;

#define B_      16
#define L_      400
#define T_      40
#define H_      256
#define E_      256
#define VOCAB_  50000
#define TWOH    512
#define THREEH  768
#define FOURH   1024
#define NTILE_V 3125          /* 50000 / 16 */
#define VBLOCKS 128
#define VWAVES  (VBLOCKS * 8) /* 1024 waves */

__device__ __forceinline__ f32x8 wmma_bf16(bf16x16 a, bf16x16 b, f32x8 c) {
  return __builtin_amdgcn_wmma_f32_16x16x32_bf16(false, a, false, b, (short)0, c,
                                                 false, false);
}

// A fragment (16x32 bf16): base points at row m=(lane&15), K = kt*32.
// lanes 0-15: K = 0..15 ; lanes 16-31: K = 16..31.
__device__ __forceinline__ bf16x16 load_a_frag(const __bf16* base, int lane) {
  int kb0 = (lane < 16) ? 0 : 16;
  return *(const bf16x16*)(base + kb0);
}

__device__ __forceinline__ float sigf(float x) { return 1.f / (1.f + expf(-x)); }

// ---------------------------------------------------------------- prep
__global__ void prep_kernel(const float* Vw, const float* Wih, const float* Whh,
                            const float* enc_states, const float* Whw,
                            const float* embed, const int* dec_input,
                            __bf16* Vwbf, __bf16* Wcat, __bf16* encbf,
                            __bf16* Whwbf, float* embT, float* cov,
                            float* lm_acc, float* covl_acc) {
  long tid = (long)blockIdx.x * blockDim.x + threadIdx.x;
  long nth = (long)gridDim.x * blockDim.x;
  for (long i = tid; i < (long)VOCAB_ * THREEH; i += nth) Vwbf[i] = (__bf16)Vw[i];
  for (long i = tid; i < (long)FOURH * TWOH; i += nth) {
    int j = (int)(i / TWOH), k = (int)(i % TWOH);
    float v = (k < H_) ? Wih[(long)j * E_ + k] : Whh[(long)j * H_ + (k - H_)];
    Wcat[i] = (__bf16)v;
  }
  for (long i = tid; i < (long)B_ * L_ * TWOH; i += nth) encbf[i] = (__bf16)enc_states[i];
  for (long i = tid; i < (long)TWOH * TWOH; i += nth) Whwbf[i] = (__bf16)Whw[i];
  for (long i = tid; i < (long)T_ * B_ * E_; i += nth) {
    int t = (int)(i / (B_ * E_));
    int r = (int)(i % (B_ * E_));
    int b = r / E_, e = r % E_;
    int row = dec_input[b * T_ + t];
    embT[i] = embed[(long)row * E_ + e];
  }
  for (long i = tid; i < B_ * L_; i += nth) cov[i] = 0.f;
  if (tid < B_) { lm_acc[tid] = 0.f; covl_acc[tid] = 0.f; }
}

// ------------------------------------------------------------ enc_proj
__global__ void __launch_bounds__(256) encproj_kernel(const __bf16* encbf,
                                                      const __bf16* Whwbf,
                                                      const float* Whb,
                                                      float* enc_proj) {
  int mtile = blockIdx.x;            // 400 tiles of 16 rows
  int wave = threadIdx.x >> 5, lane = threadIdx.x & 31;
  int mlo = lane & 15, half = lane >> 4;
  for (int nt = wave; nt < 32; nt += 8) {
    f32x8 c = {};
    const __bf16* arow = encbf + (long)(mtile * 16 + mlo) * TWOH;
    const __bf16* brow = Whwbf + (long)(nt * 16 + mlo) * TWOH + half * 16;
    for (int kt = 0; kt < 16; ++kt) {
      bf16x16 a = load_a_frag(arow + kt * 32, lane);
      bf16x16 b = *(const bf16x16*)(brow + kt * 32);
      c = wmma_bf16(a, b, c);
    }
    int n = nt * 16 + mlo;
    float bias = Whb[n];
    for (int r = 0; r < 8; ++r) {
      int m = mtile * 16 + r + half * 8;
      enc_proj[(long)m * TWOH + n] = c[r] + bias;
    }
  }
}

// ---------------------------------------------------------------- LSTM
// Persistent single-workgroup LSTM; 2-pass gates keep static LDS at 48KB.
__global__ void __launch_bounds__(1024) lstm_kernel(const float* enc_h,
                                                    const float* enc_c,
                                                    const float* embT,
                                                    const __bf16* Wcat,
                                                    const float* bih,
                                                    const float* bhh, float* hs) {
  __shared__ __bf16 Abf[B_ * TWOH];    // [b][k]: k<256 = x_t, k>=256 = h (bf16)
  __shared__ float gates[B_ * TWOH];   // two-pass gate staging (f32)
  int tid = threadIdx.x;
  int wave = tid >> 5, lane = tid & 31;
  int mlo = lane & 15, half = lane >> 4;
  float creg[4], tmp[4];
  for (int q = 0; q < 4; ++q) {
    int idx = tid + q * 1024;          // 4096 = B*H items
    int b = idx >> 8, k = idx & 255;
    creg[q] = enc_c[idx];
    Abf[b * TWOH + H_ + k] = (__bf16)enc_h[idx];
  }
  __syncthreads();
  for (int t = 0; t < T_; ++t) {
    for (int i = tid; i < B_ * E_; i += 1024) {
      int b = i >> 8, k = i & 255;
      Abf[b * TWOH + k] = (__bf16)embT[(long)t * B_ * E_ + i];
    }
    __syncthreads();
    // ---- pass 1: i (rows 0..255) and g (rows 512..767) gates, 32 tiles
    {
      int tt = wave;
      int n   = (tt < 16) ? tt * 16 + mlo : 512 + (tt - 16) * 16 + mlo;
      int col = (tt < 16) ? tt * 16 + mlo : 256 + (tt - 16) * 16 + mlo;
      f32x8 c = {};
      const __bf16* brow = Wcat + (long)n * TWOH + half * 16;
      for (int kt = 0; kt < 16; ++kt) {
        bf16x16 a = load_a_frag(&Abf[mlo * TWOH + kt * 32], lane);
        bf16x16 b = *(const bf16x16*)(brow + kt * 32);
        c = wmma_bf16(a, b, c);
      }
      float bias = bih[n] + bhh[n];
      for (int r = 0; r < 8; ++r) gates[(r + half * 8) * TWOH + col] = c[r] + bias;
    }
    __syncthreads();
    for (int q = 0; q < 4; ++q) {
      int idx = tid + q * 1024;
      int b = idx >> 8, k = idx & 255;
      float gi = gates[b * TWOH + k];
      float gg = gates[b * TWOH + 256 + k];
      tmp[q] = sigf(gi) * tanhf(gg);
    }
    __syncthreads();
    // ---- pass 2: f (rows 256..511) and o (rows 768..1023) gates
    {
      int tt = wave;
      int n   = (tt < 16) ? 256 + tt * 16 + mlo : 768 + (tt - 16) * 16 + mlo;
      int col = (tt < 16) ? tt * 16 + mlo : 256 + (tt - 16) * 16 + mlo;
      f32x8 c = {};
      const __bf16* brow = Wcat + (long)n * TWOH + half * 16;
      for (int kt = 0; kt < 16; ++kt) {
        bf16x16 a = load_a_frag(&Abf[mlo * TWOH + kt * 32], lane);
        bf16x16 b = *(const bf16x16*)(brow + kt * 32);
        c = wmma_bf16(a, b, c);
      }
      float bias = bih[n] + bhh[n];
      for (int r = 0; r < 8; ++r) gates[(r + half * 8) * TWOH + col] = c[r] + bias;
    }
    __syncthreads();
    for (int q = 0; q < 4; ++q) {
      int idx = tid + q * 1024;
      int b = idx >> 8, k = idx & 255;
      float gf = gates[b * TWOH + k];
      float go = gates[b * TWOH + 256 + k];
      float cc = sigf(gf) * creg[q] + tmp[q];
      creg[q] = cc;
      float hh = sigf(go) * tanhf(cc);
      Abf[b * TWOH + H_ + k] = (__bf16)hh;
      hs[(long)t * B_ * H_ + idx] = hh;
    }
    __syncthreads();
  }
}

// --------------------------------------------------- attention (per step)
__global__ void __launch_bounds__(512) attn_kernel(
    int t, const float* enc_states, const float* enc_proj, const float* hs,
    const float* Wsw, const float* Wsb, const float* wcw, const float* wcb,
    const float* vw, const float* vb_s, const unsigned char* enc_mask,
    const float* embT, const float* whw_v, const float* whb_s,
    const float* wsw_v, const float* wsb_s, const float* wxw_v,
    const float* wxb_s, const int* targets, const int* article_inds, float* cov,
    __bf16* Hcatbf, float* pgen, float* ptrsum, float* covl_acc) {
  int b = blockIdx.x, tid = threadIdx.x;
  __shared__ float sh_h[H_];
  __shared__ float sh_dp[TWOH];
  __shared__ float sc[L_];
  __shared__ float red[512];
  const float* ht = hs + (long)t * B_ * H_ + (long)b * H_;
  if (tid < H_) sh_h[tid] = ht[tid];
  __syncthreads();
  {  // dec_proj = h_t @ Ws_w.T + Ws_b  (one output dim per thread)
    float acc = Wsb[tid];
    const float* wr = Wsw + (long)tid * H_;
    for (int k = 0; k < H_; ++k) acc += sh_h[k] * wr[k];
    sh_dp[tid] = acc;
  }
  __syncthreads();
  int wave = tid >> 5, lane = tid & 31;
  float vbias = vb_s[0];
  for (int l = wave; l < L_; l += 16) {
    float covv = cov[b * L_ + l];
    const float* ep = enc_proj + ((long)b * L_ + l) * TWOH;
    float s = 0.f;
    for (int j = 0; j < 16; ++j) {
      int d = lane * 16 + j;
      float f = tanhf(ep[d] + sh_dp[d] + covv * wcw[d] + wcb[d]);
      s += f * vw[d];
    }
    for (int m = 16; m >= 1; m >>= 1) s += __shfl_xor(s, m, 32);
    if (lane == 0) {
      float sv = s + vbias;
      if (enc_mask[b * L_ + l]) sv = -1e30f;
      sc[l] = sv;
    }
  }
  __syncthreads();
  // softmax over L
  float lm = -INFINITY;
  for (int l = tid; l < L_; l += 512) lm = fmaxf(lm, sc[l]);
  red[tid] = lm; __syncthreads();
  for (int s2 = 256; s2 >= 1; s2 >>= 1) {
    if (tid < s2) red[tid] = fmaxf(red[tid], red[tid + s2]);
    __syncthreads();
  }
  float smax = red[0]; __syncthreads();
  float lsum = 0.f;
  for (int l = tid; l < L_; l += 512) { float e = expf(sc[l] - smax); sc[l] = e; lsum += e; }
  red[tid] = lsum; __syncthreads();
  for (int s2 = 256; s2 >= 1; s2 >>= 1) {
    if (tid < s2) red[tid] += red[tid + s2];
    __syncthreads();
  }
  float inv = 1.f / red[0]; __syncthreads();
  float covl_part = 0.f;
  for (int l = tid; l < L_; l += 512) {
    float a = sc[l] * inv;
    sc[l] = a;
    float nc = cov[b * L_ + l] + a;
    cov[b * L_ + l] = nc;
    covl_part += fminf(nc, a);
  }
  red[tid] = covl_part; __syncthreads();
  for (int s2 = 256; s2 >= 1; s2 >>= 1) {
    if (tid < s2) red[tid] += red[tid + s2];
    __syncthreads();
  }
  if (tid == 0) covl_acc[b] += red[0];
  __syncthreads();
  {  // context + Hcat(bf16) + p_gen partial
    int d = tid;
    float cx = 0.f;
    const float* es = enc_states + (long)b * L_ * TWOH + d;
    for (int l = 0; l < L_; ++l) cx += sc[l] * es[(long)l * TWOH];
    Hcatbf[(long)b * THREEH + H_ + d] = (__bf16)cx;
    float pg = cx * whw_v[d];
    if (d < H_) {
      Hcatbf[(long)b * THREEH + d] = (__bf16)sh_h[d];
      pg += sh_h[d] * wsw_v[d] + embT[(long)t * B_ * E_ + b * E_ + d] * wxw_v[d];
    }
    red[tid] = pg;
  }
  __syncthreads();
  for (int s2 = 256; s2 >= 1; s2 >>= 1) {
    if (tid < s2) red[tid] += red[tid + s2];
    __syncthreads();
  }
  if (tid == 0) pgen[b] = sigf(red[0] + whb_s[0] + wsb_s[0] + wxb_s[0]);
  __syncthreads();
  // pointer-attention mass at the target column
  int tgt = targets[b * T_ + t];
  int gidx = (tgt != 0) ? (tgt - 1) : 0;
  float ps = 0.f;
  for (int l = tid; l < L_; l += 512) {
    int idx = enc_mask[b * L_ + l] ? 0 : (article_inds[b * L_ + l] - 1);
    if (idx == gidx) ps += sc[l];
  }
  red[tid] = ps; __syncthreads();
  for (int s2 = 256; s2 >= 1; s2 >>= 1) {
    if (tid < s2) red[tid] += red[tid + s2];
    __syncthreads();
  }
  if (tid == 0) ptrsum[b] = red[0];
}

// ------------------------------------- vocab projection + online softmax
__global__ void __launch_bounds__(256) vocab_kernel(const __bf16* Vwbf,
                                                    const __bf16* Hcatbf,
                                                    const float* Vb,
                                                    float* vpart) {
  int wave = threadIdx.x >> 5, lane = threadIdx.x & 31;
  int gw = blockIdx.x * 8 + wave;
  int mlo = lane & 15, half = lane >> 4;
  float mx[8], sm[8];
  for (int r = 0; r < 8; ++r) { mx[r] = -INFINITY; sm[r] = 0.f; }
  const __bf16* arow = Hcatbf + (long)mlo * THREEH;
  for (int nt = gw; nt < NTILE_V; nt += VWAVES) {
    const __bf16* brow = Vwbf + (long)(nt * 16 + mlo) * THREEH + half * 16;
    int nn = nt + VWAVES;
    if (nn < NTILE_V)
      __builtin_prefetch(Vwbf + (long)(nn * 16 + mlo) * THREEH + half * 16, 0, 0);
    f32x8 c = {};
    for (int kt = 0; kt < 24; ++kt) {
      bf16x16 a = load_a_frag(arow + kt * 32, lane);
      bf16x16 bb = *(const bf16x16*)(brow + kt * 32);
      c = wmma_bf16(a, bb, c);
    }
    int n = nt * 16 + mlo;
    float bias = Vb[n];
    for (int r = 0; r < 8; ++r) {
      float x = c[r] + bias;
      float nm = fmaxf(mx[r], x);
      sm[r] = sm[r] * expf(mx[r] - nm) + expf(x - nm);
      mx[r] = nm;
    }
  }
  for (int m = 8; m >= 1; m >>= 1) {  // reduce within 16-lane halves
    for (int r = 0; r < 8; ++r) {
      float om = __shfl_xor(mx[r], m, 32);
      float os = __shfl_xor(sm[r], m, 32);
      float nm = fmaxf(mx[r], om);
      sm[r] = sm[r] * expf(mx[r] - nm) + os * expf(om - nm);
      mx[r] = nm;
    }
  }
  __shared__ float smax[8 * 16], ssum[8 * 16];
  if (lane == 0 || lane == 16) {
    for (int r = 0; r < 8; ++r) {
      smax[wave * 16 + half * 8 + r] = mx[r];
      ssum[wave * 16 + half * 8 + r] = sm[r];
    }
  }
  __syncthreads();
  if (threadIdx.x < 16) {
    int b = threadIdx.x;
    float m0 = smax[b], s0 = ssum[b];
    for (int w = 1; w < 8; ++w) {
      float om = smax[w * 16 + b], os = ssum[w * 16 + b];
      float nm = fmaxf(m0, om);
      s0 = s0 * expf(m0 - nm) + os * expf(om - nm);
      m0 = nm;
    }
    vpart[((long)blockIdx.x * 16 + b) * 2 + 0] = m0;
    vpart[((long)blockIdx.x * 16 + b) * 2 + 1] = s0;
  }
}

// ------------------------------------------------------ combine per step
__global__ void __launch_bounds__(256) combine_kernel(
    int t, const float* vpart, const __bf16* Hcatbf, const __bf16* Vwbf,
    const float* Vb, const int* targets, const float* pgen,
    const float* ptrsum, float* lm_acc) {
  __shared__ float gmax[16], gsum[16];
  __shared__ float red[256];
  int tid = threadIdx.x;
  if (tid < 16) {
    int b = tid;
    float m0 = vpart[b * 2], s0 = vpart[b * 2 + 1];
    for (int k = 1; k < VBLOCKS; ++k) {
      float om = vpart[((long)k * 16 + b) * 2], os = vpart[((long)k * 16 + b) * 2 + 1];
      float nm = fmaxf(m0, om);
      s0 = s0 * expf(m0 - nm) + os * expf(om - nm);
      m0 = nm;
    }
    gmax[b] = m0; gsum[b] = s0;
  }
  __syncthreads();
  int b = tid >> 4, j = tid & 15;
  int tgt = targets[b * T_ + t];
  int gidx = (tgt != 0) ? (tgt - 1) : 0;
  float part = 0.f;
  const __bf16* hb = Hcatbf + (long)b * THREEH;
  const __bf16* vr = Vwbf + (long)gidx * THREEH;
  for (int k = j; k < THREEH; k += 16) part += (float)hb[k] * (float)vr[k];
  red[tid] = part;
  __syncthreads();
  for (int s2 = 8; s2 >= 1; s2 >>= 1) {
    if (j < s2) red[tid] += red[tid + s2];
    __syncthreads();
  }
  if (j == 0) {
    float lt = red[b * 16] + Vb[gidx];
    float pv = expf(lt - gmax[b]) / gsum[b];
    float pg = pgen[b];
    float out = pg * pv + (1.f - pg) * ptrsum[b];
    if (tgt != 0) lm_acc[b] += -logf(out);
  }
}

// --------------------------------------------------------------- finalize
__global__ void finalize_kernel(const int* dec_input, const float* lm_acc,
                                const float* covl_acc, float* out) {
  int b = threadIdx.x;
  if (b < B_) {
    int cnt = 0;
    for (int t = 0; t < T_; ++t) cnt += (dec_input[b * T_ + t] > 0) ? 1 : 0;
    float dl = (float)cnt;
    out[b] = lm_acc[b] / dl + 1.0f * covl_acc[b] / dl;
  }
}

extern "C" void kernel_launch(void* const* d_in, const int* in_sizes, int n_in,
                              void* d_out, int out_size, void* d_ws,
                              size_t ws_size, hipStream_t stream) {
  (void)in_sizes; (void)n_in; (void)out_size; (void)ws_size;
  const float* enc_states = (const float*)d_in[0];
  const float* enc_h  = (const float*)d_in[1];
  const float* enc_c  = (const float*)d_in[2];
  const float* embed  = (const float*)d_in[3];
  const float* Wih    = (const float*)d_in[4];
  const float* Whh    = (const float*)d_in[5];
  const float* bih    = (const float*)d_in[6];
  const float* bhh    = (const float*)d_in[7];
  const float* Whw    = (const float*)d_in[8];
  const float* Whb    = (const float*)d_in[9];
  const float* Wsw    = (const float*)d_in[10];
  const float* Wsb    = (const float*)d_in[11];
  const float* wcw    = (const float*)d_in[12];
  const float* wcb    = (const float*)d_in[13];
  const float* vw     = (const float*)d_in[14];
  const float* vb     = (const float*)d_in[15];
  const float* whwv   = (const float*)d_in[16];
  const float* whb    = (const float*)d_in[17];
  const float* wswv   = (const float*)d_in[18];
  const float* wsb    = (const float*)d_in[19];
  const float* wxwv   = (const float*)d_in[20];
  const float* wxb    = (const float*)d_in[21];
  const float* Vw     = (const float*)d_in[22];
  const float* Vb     = (const float*)d_in[23];
  const int* dec_input    = (const int*)d_in[24];
  const int* targets      = (const int*)d_in[25];
  const int* article_inds = (const int*)d_in[26];
  const unsigned char* enc_mask = (const unsigned char*)d_in[27];

  char* ws = (char*)d_ws;
  size_t off = 0;
  auto alloc = [&](size_t bytes) {
    void* p = ws + off;
    off += bytes;
    off = (off + 255) & ~(size_t)255;
    return p;
  };
  __bf16* Vwbf    = (__bf16*)alloc((size_t)VOCAB_ * THREEH * 2);
  __bf16* Wcat    = (__bf16*)alloc((size_t)FOURH * TWOH * 2);
  __bf16* encbf   = (__bf16*)alloc((size_t)B_ * L_ * TWOH * 2);
  __bf16* Whwbf   = (__bf16*)alloc((size_t)TWOH * TWOH * 2);
  float* enc_proj = (float*)alloc((size_t)B_ * L_ * TWOH * 4);
  float* embT     = (float*)alloc((size_t)T_ * B_ * E_ * 4);
  float* hs       = (float*)alloc((size_t)T_ * B_ * H_ * 4);
  float* cov      = (float*)alloc((size_t)B_ * L_ * 4);
  __bf16* Hcatbf  = (__bf16*)alloc((size_t)B_ * THREEH * 2);
  float* vpart    = (float*)alloc((size_t)VBLOCKS * 16 * 2 * 4);
  float* pgen     = (float*)alloc(16 * 4);
  float* ptrsum   = (float*)alloc(16 * 4);
  float* lm_acc   = (float*)alloc(16 * 4);
  float* covl_acc = (float*)alloc(16 * 4);

  prep_kernel<<<2048, 256, 0, stream>>>(Vw, Wih, Whh, enc_states, Whw, embed,
                                        dec_input, Vwbf, Wcat, encbf, Whwbf,
                                        embT, cov, lm_acc, covl_acc);
  encproj_kernel<<<400, 256, 0, stream>>>(encbf, Whwbf, Whb, enc_proj);
  lstm_kernel<<<1, 1024, 0, stream>>>(enc_h, enc_c, embT, Wcat, bih, bhh, hs);
  for (int t = 0; t < T_; ++t) {
    attn_kernel<<<16, 512, 0, stream>>>(
        t, enc_states, enc_proj, hs, Wsw, Wsb, wcw, wcb, vw, vb, enc_mask,
        embT, whwv, whb, wswv, wsb, wxwv, wxb, targets, article_inds, cov,
        Hcatbf, pgen, ptrsum, covl_acc);
    vocab_kernel<<<VBLOCKS, 256, 0, stream>>>(Vwbf, Hcatbf, Vb, vpart);
    combine_kernel<<<1, 256, 0, stream>>>(t, vpart, Hcatbf, Vwbf, Vb, targets,
                                          pgen, ptrsum, lm_acc);
  }
  finalize_kernel<<<1, 64, 0, stream>>>(dec_input, lm_acc, covl_acc,
                                        (float*)d_out);
}
